// LSTMnoTF_9577777070394
// MI455X (gfx1250) — matrix operand order repaired and checked by reading, not compiled, not run on hardware
//
#include <hip/hip_runtime.h>
#include <math.h>

typedef __attribute__((ext_vector_type(16))) _Float16     v16h;
typedef __attribute__((ext_vector_type(8)))  _Float16     v8h;
typedef __attribute__((ext_vector_type(8)))  float        v8f;
typedef __attribute__((ext_vector_type(4)))  unsigned int v4u;

#define AS1 __attribute__((address_space(1)))   // global
#define AS3 __attribute__((address_space(3)))   // LDS

typedef AS1 const _Float16* gcp_h;   // pointer to global const half
typedef AS1 _Float16*       gp_h;    // pointer to global half
typedef AS1 float*          gp_f;
typedef AS1 const float*    gcp_f;
typedef AS3 const _Float16* lcp_h;   // pointer to LDS const half
typedef AS3 _Float16*       lp_h;

#define B_ 256
#define T_ 256
#define I_ 32
#define H_ 512
#define O_ 32
#define F_ 128

#define NBLK   32            // one block per hidden-column tile (H/16)
#define NTHR   512           // 16 waves: one per batch-row tile (B/16)
#define LPITCH 520           // 512 halves + 8 pad halves to spread LDS banks

// ---------------------------------------------------------------------------
// math helpers (v_exp_f32 transcendental; tanh saturates correctly)
// ---------------------------------------------------------------------------
__device__ __forceinline__ float sigf(float x) {
    return 1.0f / (1.0f + __expf(-x));
}
__device__ __forceinline__ float tanh1(float x) {
    float e = __expf(2.0f * x);
    return 1.0f - 2.0f / (e + 1.0f);
}

// ---------------------------------------------------------------------------
// CDNA5 16-bit A-operand lane layout (16x32 MxK, wave32), contiguous K to match B:
//   lanes 0-15  : K=[0..15] in v0-7   (row M = lane)
//   lanes 16-31 : K=[16..31] in v0-7  (row M = lane-16)
// -> per lane: two contiguous 16B global loads at k0+lh*16 and k0+lh*16+8.
// ---------------------------------------------------------------------------
__device__ __forceinline__ v16h load_a16g(gcp_h rowk, int lh) {
    v8h lo = *(AS1 const v8h*)(rowk + lh * 16);
    v8h hi = *(AS1 const v8h*)(rowk + lh * 16 + 8);
    return __builtin_shufflevector(lo, hi, 0, 1, 2, 3, 4, 5, 6, 7,
                                           8, 9, 10, 11, 12, 13, 14, 15);
}
// B-operand (32x16 KxN): lane l holds column N=l%16, K=(l/16)*16..+15 contiguous.
// Since B = W^T, that is a contiguous 16-half slice of a row of W.
__device__ __forceinline__ v16h load_b16g(gcp_h rowk, int lh) {
    v8h lo = *(AS1 const v8h*)(rowk + lh * 16);
    v8h hi = *(AS1 const v8h*)(rowk + lh * 16 + 8);
    return __builtin_shufflevector(lo, hi, 0, 1, 2, 3, 4, 5, 6, 7,
                                           8, 9, 10, 11, 12, 13, 14, 15);
}
// same B-operand layout but from LDS: two ds_load_b128 per operand
__device__ __forceinline__ v16h load_b16l(lcp_h rowk, int lh) {
    v8h lo = *(AS3 const v8h*)(rowk + lh * 16);
    v8h hi = *(AS3 const v8h*)(rowk + lh * 16 + 8);
    return __builtin_shufflevector(lo, hi, 0, 1, 2, 3, 4, 5, 6, 7,
                                           8, 9, 10, 11, 12, 13, 14, 15);
}

// ---------------------------------------------------------------------------
// device-wide sense barrier (32 blocks, <=67KB LDS each on 320KB/WGP: all
// blocks trivially co-resident)
// ---------------------------------------------------------------------------
__device__ __forceinline__ void grid_sync(unsigned* cnt, unsigned* gen, unsigned nblk) {
    __threadfence();
    __syncthreads();
    if (threadIdx.x == 0) {
        unsigned g = __hip_atomic_load(gen, __ATOMIC_RELAXED, __HIP_MEMORY_SCOPE_AGENT);
        unsigned prev = __hip_atomic_fetch_add(cnt, 1u, __ATOMIC_ACQ_REL, __HIP_MEMORY_SCOPE_AGENT);
        if (prev == nblk - 1u) {
            __hip_atomic_store(cnt, 0u, __ATOMIC_RELAXED, __HIP_MEMORY_SCOPE_AGENT);
            __hip_atomic_fetch_add(gen, 1u, __ATOMIC_RELEASE, __HIP_MEMORY_SCOPE_AGENT);
        } else {
            while (__hip_atomic_load(gen, __ATOMIC_ACQUIRE, __HIP_MEMORY_SCOPE_AGENT) == g) {
                __builtin_amdgcn_s_sleep(2);
            }
        }
    }
    __syncthreads();
    __threadfence();
}

// ---------------------------------------------------------------------------
// init: f32->f16 conversion, bias fold, zero h0 and barrier state
// ---------------------------------------------------------------------------
__global__ void lstm_init_kernel(const float* __restrict__ x,
                                 const float* __restrict__ Wih,
                                 const float* __restrict__ Whh,
                                 const float* __restrict__ bih,
                                 const float* __restrict__ bhh,
                                 const float* __restrict__ Wfc,
                                 _Float16* __restrict__ x16,
                                 _Float16* __restrict__ Wih16,
                                 _Float16* __restrict__ Whh16,
                                 _Float16* __restrict__ Wfc16,
                                 float* __restrict__ bias,
                                 _Float16* __restrict__ h0,
                                 unsigned* __restrict__ bar) {
    size_t tid = (size_t)blockIdx.x * blockDim.x + threadIdx.x;
    size_t stride = (size_t)gridDim.x * blockDim.x;
    for (size_t i = tid; i < (size_t)B_ * T_ * I_; i += stride) x16[i] = (_Float16)x[i];
    for (size_t i = tid; i < (size_t)4 * H_ * I_; i += stride) Wih16[i] = (_Float16)Wih[i];
    for (size_t i = tid; i < (size_t)4 * H_ * H_; i += stride) Whh16[i] = (_Float16)Whh[i];
    for (size_t i = tid; i < (size_t)O_ * H_; i += stride) Wfc16[i] = (_Float16)Wfc[i];
    for (size_t i = tid; i < (size_t)4 * H_; i += stride) bias[i] = bih[i] + bhh[i];
    for (size_t i = tid; i < (size_t)B_ * H_; i += stride) h0[i] = (_Float16)0.0f;
    if (tid < 2) bar[tid] = 0u;
}

// ---------------------------------------------------------------------------
// persistent LSTM: the whole encoder+decoder in one launch.
// block b = hidden tile nt (n0 = 16*b); wave w = batch tile mt (m0 = 16*w).
// W_hh slice lives in LDS for the whole kernel (ds_load B-operands);
// W_ih B-tiles, bias and the c tile live in registers for the whole kernel.
// Per step only h ping-pongs through global (L2-resident).
// ---------------------------------------------------------------------------
__global__ __launch_bounds__(NTHR) void lstm_persistent(
    const _Float16* __restrict__ x16_,   // [B, T, I]
    const _Float16* __restrict__ Whh_,   // [4H, H]
    const _Float16* __restrict__ Wih_,   // [4H, I]
    const float* __restrict__ bias_,     // [4H]
    const _Float16* __restrict__ Wfc_,   // [O, H]
    const float* __restrict__ bfc_,      // [O]
    _Float16* __restrict__ hb0_,         // [B, H] ping
    _Float16* __restrict__ hb1_,         // [B, H] pong
    _Float16* __restrict__ pred16_,      // [B, O]
    float* __restrict__ out_,            // [B, F, O]
    unsigned* __restrict__ bar)          // [2]: counter, generation
{
    extern __shared__ char smem_raw[];
    lp_h lwhh = (lp_h)smem_raw;                    // [4][16][LPITCH] halves

    // explicit global address space for all hot pointers
    gcp_h x16    = (gcp_h)x16_;
    gcp_h Whh    = (gcp_h)Whh_;
    gcp_h Wih    = (gcp_h)Wih_;
    gcp_f bias   = (gcp_f)bias_;
    gcp_h Wfc    = (gcp_h)Wfc_;
    gcp_f bfc    = (gcp_f)bfc_;
    gp_h  pred16 = (gp_h)pred16_;
    gp_f  out    = (gp_f)out_;

    const int lane = (int)(threadIdx.x & 31);
    const int w    = (int)(threadIdx.x >> 5);      // 0..15 -> mt
    const int n0   = (int)blockIdx.x * 16;
    const int m0   = w * 16;
    const int lm   = lane & 15, lh = lane >> 4;

    // ---- stage this block's W_hh slice into LDS (rows g*H + n0 .. +16) ----
    for (int i = (int)threadIdx.x; i < 4 * 16 * (H_ / 8); i += NTHR) {
        int half8 = i * 8;                          // 8 halves per copy
        int row   = half8 >> 9;                     // /512
        int col   = half8 & 511;
        int g = row >> 4, rr = row & 15;
        *(AS3 v4u*)(lwhh + (g * 16 + rr) * LPITCH + col) =
            *(AS1 const v4u*)(Whh + ((size_t)(g * H_ + n0 + rr)) * H_ + col);
    }
    __syncthreads();

    // ---- persistent per-wave state ----
    v16h wih_t[4];
    float bias_r[4];
#pragma unroll
    for (int g = 0; g < 4; ++g) {
        wih_t[g]  = load_b16g(Wih + (size_t)(g * H_ + n0 + lm) * I_, lh);
        bias_r[g] = bias[g * H_ + n0 + lm];
    }
    float c_reg[8];
#pragma unroll
    for (int r = 0; r < 8; ++r) c_reg[r] = 0.0f;

    for (int step = 0; step < T_ + F_; ++step) {
        gcp_h hin  = (step & 1) ? (gcp_h)hb1_ : (gcp_h)hb0_;
        gp_h  hout = (step & 1) ? (gp_h)hb0_  : (gp_h)hb1_;

        gcp_h xin;
        int xstride;
        if (step < T_) {
            xin = x16 + (size_t)step * I_;
            xstride = T_ * I_;
        } else {
            // ---- decoder: blocks 0,1 produce pred = h @ Wfc.T + bfc ----
            const int k = step - T_;
            if (blockIdx.x < 2) {
                const int po0 = (int)blockIdx.x * 16;    // output-col tile of O
                v8f pacc;
                {
                    float bv = bfc[po0 + lm];
#pragma unroll
                    for (int r = 0; r < 8; ++r) pacc[r] = bv;
                }
#pragma unroll 4
                for (int k0 = 0; k0 < H_; k0 += 32) {
                    v16h a = load_a16g(hin + (size_t)(m0 + lm) * H_ + k0, lh);
                    v16h b = load_b16g(Wfc + (size_t)(po0 + lm) * H_ + k0, lh);
                    pacc = __builtin_amdgcn_wmma_f32_16x16x32_f16(
                        false, a, false, b, (short)0, pacc, false, false);
                }
#pragma unroll
                for (int r = 0; r < 8; ++r) {
                    const int m = m0 + r + lh * 8;
                    const int n = po0 + lm;
                    float v = pacc[r];
                    out[(size_t)m * (F_ * O_) + (size_t)k * O_ + n] = v;
                    pred16[(size_t)m * O_ + n] = (_Float16)v;
                }
            }
            grid_sync(&bar[0], &bar[1], NBLK);       // pred visible everywhere
            xin = (gcp_h)pred16;
            xstride = O_;
        }

        // ---- gates = x @ Wih.T + h @ Whh.T + b ----
        v8f acc[4];
#pragma unroll
        for (int g = 0; g < 4; ++g) {
#pragma unroll
            for (int r = 0; r < 8; ++r) acc[g][r] = bias_r[g];
        }
        {   // input contribution (K = 32, register-resident weights)
            v16h a = load_a16g(xin + (size_t)(m0 + lm) * xstride, lh);
#pragma unroll
            for (int g = 0; g < 4; ++g)
                acc[g] = __builtin_amdgcn_wmma_f32_16x16x32_f16(
                    false, a, false, wih_t[g], (short)0, acc[g], false, false);
        }
        // recurrent: A from global h (global_load), B from LDS W_hh (ds_load)
#pragma unroll 4
        for (int k0 = 0; k0 < H_; k0 += 32) {
            v16h a = load_a16g(hin + (size_t)(m0 + lm) * H_ + k0, lh);
#pragma unroll
            for (int g = 0; g < 4; ++g) {
                v16h b = load_b16l(lwhh + (g * 16 + lm) * LPITCH + k0, lh);
                acc[g] = __builtin_amdgcn_wmma_f32_16x16x32_f16(
                    false, a, false, b, (short)0, acc[g], false, false);
            }
        }

        // ---- LSTM cell on accumulators; c stays in registers ----
        // C/D layout: vgpr r -> M = m0 + r + 8*lh, N = n0 + lm
#pragma unroll
        for (int r = 0; r < 8; ++r) {
            const int m = m0 + r + lh * 8;
            const int n = n0 + lm;
            float iv = sigf(acc[0][r]);
            float fv = sigf(acc[1][r]);
            float gv = tanh1(acc[2][r]);
            float ov = sigf(acc[3][r]);
            float cn = fv * c_reg[r] + iv * gv;
            c_reg[r] = cn;
            hout[(size_t)m * H_ + n] = (_Float16)(ov * tanh1(cn));
        }

        grid_sync(&bar[0], &bar[1], NBLK);           // h_out visible everywhere
    }
}

// ---------------------------------------------------------------------------
// host launcher: 2 launches total
// ---------------------------------------------------------------------------
extern "C" void kernel_launch(void* const* d_in, const int* in_sizes, int n_in,
                              void* d_out, int out_size, void* d_ws, size_t ws_size,
                              hipStream_t stream) {
    (void)in_sizes; (void)n_in; (void)out_size; (void)ws_size;

    const float* x   = (const float*)d_in[0];  // [B,T,I]
    const float* Wih = (const float*)d_in[1];  // [4H,I]
    const float* Whh = (const float*)d_in[2];  // [4H,H]
    const float* bih = (const float*)d_in[3];  // [4H]
    const float* bhh = (const float*)d_in[4];  // [4H]
    const float* Wfc = (const float*)d_in[5];  // [O,H]
    const float* bfc = (const float*)d_in[6];  // [O]
    float* out = (float*)d_out;                // [B,F,O]

    char* ws = (char*)d_ws;
    auto carve = [&](size_t bytes) -> char* {
        char* p = ws;
        ws += (bytes + 255) & ~(size_t)255;
        return p;
    };
    _Float16* x16    = (_Float16*)carve((size_t)B_ * T_ * I_ * 2);   // 4 MB
    _Float16* Wih16  = (_Float16*)carve((size_t)4 * H_ * I_ * 2);    // 128 KB
    _Float16* Whh16  = (_Float16*)carve((size_t)4 * H_ * H_ * 2);    // 2 MB
    _Float16* Wfc16  = (_Float16*)carve((size_t)O_ * H_ * 2);        // 32 KB
    float*    bias   = (float*)carve((size_t)4 * H_ * 4);            // 8 KB
    _Float16* hbuf0  = (_Float16*)carve((size_t)B_ * H_ * 2);        // 256 KB
    _Float16* hbuf1  = (_Float16*)carve((size_t)B_ * H_ * 2);        // 256 KB
    _Float16* pred16 = (_Float16*)carve((size_t)B_ * O_ * 2);        // 16 KB
    unsigned* bar    = (unsigned*)carve(256);

    lstm_init_kernel<<<1024, 256, 0, stream>>>(x, Wih, Whh, bih, bhh, Wfc,
                                               x16, Wih16, Whh16, Wfc16,
                                               bias, hbuf0, bar);

    const size_t lds_bytes = (size_t)4 * 16 * LPITCH * sizeof(_Float16); // 66,560 B
    lstm_persistent<<<NBLK, NTHR, lds_bytes, stream>>>(
        x16, Whh16, Wih16, bias, Wfc16, bfc,
        hbuf0, hbuf1, pred16, out, bar);
}
